// MoEModel_15444702396744
// MI455X (gfx1250) — matrix operand order repaired and checked by reading, not compiled, not run on hardware
//
#include <hip/hip_runtime.h>

typedef __attribute__((ext_vector_type(16))) _Float16 v16h;
typedef __attribute__((ext_vector_type(8)))  float    v8f;

#define BATCH 1024
#define IC 3
#define HW 32
#define NPIX 1024      // 32*32
#define NEXP 8
#define NC 100
#define GCH 16
#define ECH 32
#define NCH 272        // NEXP*ECH + GCH  (17 tiles of 16, no N padding)
#define KDIM 27        // IC*3*3
#define KPAD 32        // one f16 WMMA K-step
#define HALF_PIX 512

// ---------------------------------------------------------------------------
// Kernel 1: fused (gate + all-expert) conv as im2col GEMM on WMMA f16,
// with bias+ReLU+mean-pool epilogue. One workgroup (8 wave32) per image.
// ---------------------------------------------------------------------------
__global__ __launch_bounds__(256)
void moe_conv_pool_kernel(const float* __restrict__ x,
                          const float* __restrict__ gate_conv_w,
                          const float* __restrict__ gate_conv_b,
                          const float* __restrict__ exp_conv_w,
                          const float* __restrict__ exp_conv_b,
                          float* __restrict__ pooled) {
  __shared__ _Float16 sA[HALF_PIX * KPAD];   // 32 KB: im2col'd pixels (half image)
  __shared__ _Float16 sB[NCH * KPAD];        // 17 KB: fused weights, k-contiguous
  __shared__ float    sBias[NCH];
  __shared__ float    sPool[NCH];

  const int b    = blockIdx.x;
  const int tid  = threadIdx.x;
  const int lane = tid & 31;
  const int wave = tid >> 5;

  // ---- Stage fused weight matrix (K padded to 32 with zeros) + bias ----
  for (int i = tid; i < NCH * KPAD; i += 256) {
    const int n = i >> 5;
    const int k = i & 31;
    float wv = 0.0f;
    if (k < KDIM) {
      const int ci  = k / 9;
      const int rem = k % 9;
      if (n < NEXP * ECH)
        wv = exp_conv_w[(n * IC + ci) * 9 + rem];                 // [E,ECH,IC,3,3]
      else
        wv = gate_conv_w[((n - NEXP * ECH) * IC + ci) * 9 + rem]; // [GCH,IC,3,3]
    }
    sB[n * KPAD + k] = (_Float16)wv;
  }
  for (int i = tid; i < NCH; i += 256) {
    sBias[i] = (i < NEXP * ECH) ? exp_conv_b[i] : gate_conv_b[i - NEXP * ECH];
    sPool[i] = 0.0f;
  }

  const float* __restrict__ xb = x + (size_t)b * IC * NPIX;

  for (int half = 0; half < 2; ++half) {
    __syncthreads();  // sB/sPool ready; sA safe to overwrite

    // ---- im2col: one pixel per thread-iteration; registers -> 4x b128 stores
    for (int mm = tid; mm < HALF_PIX; mm += 256) {
      const int p = half * HALF_PIX + mm;
      const int h = p >> 5;
      const int w = p & 31;
      union { _Float16 hh[KPAD]; uint4 q[4]; } row;
      #pragma unroll
      for (int k = 0; k < KPAD; ++k) row.hh[k] = (_Float16)0.0f;  // pad + borders
      #pragma unroll
      for (int ci = 0; ci < IC; ++ci) {
        #pragma unroll
        for (int r = 0; r < 3; ++r) {
          const int hr = h + r - 1;
          if (hr >= 0 && hr < HW) {
            const float* __restrict__ src = xb + (ci * HW + hr) * HW;
            #pragma unroll
            for (int c = 0; c < 3; ++c) {
              const int wc = w + c - 1;
              if (wc >= 0 && wc < HW)
                row.hh[ci * 9 + r * 3 + c] = (_Float16)src[wc];
            }
          }
        }
      }
      uint4* dst = (uint4*)(sA + mm * KPAD);
      #pragma unroll
      for (int q = 0; q < 4; ++q) dst[q] = row.q[q];
    }
    __syncthreads();

    // ---- GEMM 512 x 272 x 32. Work item = (n-tile, 16 m-tiles).
    // B fragment, bias and pool accumulator live in registers across the
    // m-loop; A fragment is double-buffered to overlap ds loads with WMMA.
    for (int wi = wave; wi < 17 * 2; wi += 8) {
      const int nt   = wi >> 1;        // 0..16
      const int mg   = wi & 1;         // which group of 16 m-tiles
      const int ncol = nt * 16 + (lane & 15);

      union AF { v16h v; unsigned int u[8]; };

      AF ub;   // B fragment: lanes 0-15 K=0..7,16..23, lanes 16-31 K=8..15,24..31, N=lane&15
      {
        const unsigned int* pB = (const unsigned int*)(sB + ncol * KPAD);
        const int kb = (lane >> 4) * 4;
        #pragma unroll
        for (int j = 0; j < 8; ++j) ub.u[j] = pB[kb + ((j >> 2) * 8) + (j & 3)];
      }
      const float bias = sBias[ncol];
      v8f paccv = {0.f, 0.f, 0.f, 0.f, 0.f, 0.f, 0.f, 0.f};

      // A base (uint units): mrow = (mg*16 + mt0)*16 + (lane&15), *16 uints/row
      const unsigned int* pA =
          (const unsigned int*)sA + (mg * 256 + (lane & 15)) * (KPAD / 2);
      const int koff = (lane >> 4) * 4;   // lanes 16-31 shift K by +8 halves

      AF frag[2];
      #pragma unroll
      for (int j = 0; j < 8; ++j)
        frag[0].u[j] = pA[koff + ((j >> 2) * 8) + (j & 3)];

      #pragma unroll
      for (int mt0 = 0; mt0 < 16; ++mt0) {
        const int cur = mt0 & 1;
        if (mt0 < 15) {   // prefetch next A fragment
          const unsigned int* pN = pA + (mt0 + 1) * 256;
          #pragma unroll
          for (int j = 0; j < 8; ++j)
            frag[cur ^ 1].u[j] = pN[koff + ((j >> 2) * 8) + (j & 3)];
        }
        v8f acc = {0.f, 0.f, 0.f, 0.f, 0.f, 0.f, 0.f, 0.f};
        acc = __builtin_amdgcn_wmma_f32_16x16x32_f16(false, frag[cur].v,
                                                     false, ub.v,
                                                     (short)0, acc, false, false);
        #pragma unroll
        for (int r2 = 0; r2 < 8; ++r2) {
          const float vv = acc[r2] + bias;
          paccv[r2] += vv > 0.0f ? vv : 0.0f;
        }
      }
      float pacc = (((paccv[0] + paccv[1]) + (paccv[2] + paccv[3])) +
                    ((paccv[4] + paccv[5]) + (paccv[6] + paccv[7])));
      atomicAdd(&sPool[ncol], pacc);
    }
  }

  __syncthreads();
  for (int i = tid; i < NCH; i += 256)
    pooled[(size_t)b * NCH + i] = sPool[i] * (1.0f / (float)NPIX);
}

// ---------------------------------------------------------------------------
// Kernel 2: gate linear + softmax + argmax + selected-expert head.
// One block of 128 threads per image (negligible FLOPs).
// ---------------------------------------------------------------------------
__global__ __launch_bounds__(128)
void moe_router_head_kernel(const float* __restrict__ pooled,
                            const float* __restrict__ gate_lin_w,
                            const float* __restrict__ gate_lin_b,
                            const float* __restrict__ exp_lin_w,
                            const float* __restrict__ exp_lin_b,
                            float* __restrict__ out) {
  __shared__ float sg[GCH];
  __shared__ float slog[NEXP];
  __shared__ float sw[NEXP];
  __shared__ int   sbest;
  __shared__ float sp[ECH];

  const int b = blockIdx.x;
  const int t = threadIdx.x;
  const float* pb = pooled + (size_t)b * NCH;

  if (t < GCH) sg[t] = pb[NEXP * ECH + t];
  __syncthreads();

  if (t < NEXP) {
    float a = gate_lin_b[t];
    #pragma unroll
    for (int c = 0; c < GCH; ++c) a += sg[c] * gate_lin_w[t * GCH + c];
    slog[t] = a;
  }
  __syncthreads();

  if (t == 0) {
    float mx = slog[0];
    for (int e = 1; e < NEXP; ++e) mx = slog[e] > mx ? slog[e] : mx;
    float ssum = 0.0f;
    for (int e = 0; e < NEXP; ++e) { const float ex = __expf(slog[e] - mx); sw[e] = ex; ssum += ex; }
    const float inv = 1.0f / ssum;
    for (int e = 0; e < NEXP; ++e) sw[e] *= inv;
    int best = 0; float bv = sw[0];
    for (int e = 1; e < NEXP; ++e) if (sw[e] > bv) { bv = sw[e]; best = e; } // first-max tiebreak
    sbest = best;
  }
  __syncthreads();

  if (t < NEXP) out[(size_t)BATCH * NC + (size_t)b * NEXP + t] = sw[t];

  const int best = sbest;
  if (t < ECH) sp[t] = pb[best * ECH + t];
  __syncthreads();

  if (t < NC) {
    float a = exp_lin_b[best * NC + t];
    const float* wrow = exp_lin_w + ((size_t)best * NC + t) * ECH;
    #pragma unroll
    for (int c = 0; c < ECH; ++c) a += sp[c] * wrow[c];
    out[(size_t)b * NC + t] = a;
  }
}

// ---------------------------------------------------------------------------
extern "C" void kernel_launch(void* const* d_in, const int* in_sizes, int n_in,
                              void* d_out, int out_size, void* d_ws, size_t ws_size,
                              hipStream_t stream) {
  (void)in_sizes; (void)n_in; (void)out_size; (void)ws_size;
  const float* x           = (const float*)d_in[0];
  const float* gate_conv_w = (const float*)d_in[1];
  const float* gate_conv_b = (const float*)d_in[2];
  const float* gate_lin_w  = (const float*)d_in[3];
  const float* gate_lin_b  = (const float*)d_in[4];
  const float* exp_conv_w  = (const float*)d_in[5];
  const float* exp_conv_b  = (const float*)d_in[6];
  const float* exp_lin_w   = (const float*)d_in[7];
  const float* exp_lin_b   = (const float*)d_in[8];
  float* out    = (float*)d_out;
  float* pooled = (float*)d_ws;  // [BATCH, 272] fp32, ~1.1 MB

  moe_conv_pool_kernel<<<BATCH, 256, 0, stream>>>(
      x, gate_conv_w, gate_conv_b, exp_conv_w, exp_conv_b, pooled);
  moe_router_head_kernel<<<BATCH, 128, 0, stream>>>(
      pooled, gate_lin_w, gate_lin_b, exp_lin_w, exp_lin_b, out);
}
